// SpiralDeblock_68607807586563
// MI455X (gfx1250) — matrix operand order repaired and checked by reading, not compiled, not run on hardware
//
#include <hip/hip_runtime.h>
#include <hip/hip_bf16.h>
#include <stdint.h>

// Problem constants (match reference)
#define B_      16
#define N_IN    7000
#define N_OUT   28000
#define C_IN    64
#define C_OUT   64
#define S_      9
#define NNZ     (3 * N_OUT)       // 84000
#define KTOT    (S_ * C_IN)       // 576
#define KSTEPS  (KTOT / 32)       // 18 WMMA k-steps
#define NTILES  (N_OUT / 16)      // 1750
#define OTILES  (C_OUT / 16)      // 4

typedef __bf16 v16bf __attribute__((ext_vector_type(16)));
typedef float  v8f   __attribute__((ext_vector_type(8)));

static __device__ __forceinline__ uint32_t f32_to_bf16_rne(float f) {
    uint32_t u = __float_as_uint(f);
    uint32_t r = u + 0x7FFFu + ((u >> 16) & 1u);
    return r >> 16;
}

// ---------------- Kernel 1: zero the fp32 pooled accumulator ----------------
__global__ __launch_bounds__(256) void zero_f32_kernel(float4* __restrict__ p, int n4) {
    int i = blockIdx.x * 256 + threadIdx.x;
    if (i < n4) p[i] = make_float4(0.f, 0.f, 0.f, 0.f);
}

// ---------------- Kernel 2: edge scatter-add (fp32 atomics) -----------------
// thread = (edge, channel); loops over batch. x and pooled are L2-resident.
__global__ __launch_bounds__(256) void scatter_kernel(const float* __restrict__ x,
                                                      const float* __restrict__ values,
                                                      const int*   __restrict__ row,
                                                      const int*   __restrict__ col,
                                                      float*       __restrict__ pooled) {
    int gid = blockIdx.x * 256 + threadIdx.x;   // NNZ*64 threads
    int e = gid >> 6;
    int c = gid & 63;
    if (e >= NNZ) return;
    float v  = values[e];
    int   rr = row[e];
    int   cc = col[e];
    size_t xo = (size_t)cc * C_IN + c;
    size_t po = (size_t)rr * C_IN + c;
    #pragma unroll
    for (int b = 0; b < B_; ++b) {
        float g = x[(size_t)b * N_IN * C_IN + xo] * v;
        atomicAdd(&pooled[(size_t)b * N_OUT * C_IN + po], g);
    }
}

// ---------------- Kernel 3: pooled fp32 -> bf16 (one-time, avoids 9x cvt) ---
__global__ __launch_bounds__(256) void cvt_bf16_kernel(const float2* __restrict__ pf,
                                                       uint32_t*     __restrict__ pb,
                                                       int n2) {
    int i = blockIdx.x * 256 + threadIdx.x;
    if (i < n2) {
        float2 f = pf[i];
        pb[i] = f32_to_bf16_rne(f.x) | (f32_to_bf16_rne(f.y) << 16);
    }
}

// ---------------- Kernel 4: pre-swizzle weight into B-fragment layout -------
// B-matrix 32x16 bf16 layout (ISA 7.12.2): lane holds column N=lane%16,
// VGPR j packs K = kb+2j (lo), kb+2j+1 (hi), kb = 8*(lane>=16) + 8*(j>=4).
// wf[((ot*KSTEPS + t)*32 + lane)*8 + j]
__global__ __launch_bounds__(256) void wfrag_kernel(const float* __restrict__ w,
                                                    uint32_t*    __restrict__ wf) {
    int gid = blockIdx.x * 256 + threadIdx.x;       // OTILES*KSTEPS*32*8 = 18432
    if (gid >= OTILES * KSTEPS * 32 * 8) return;
    int j    = gid & 7;
    int lane = (gid >> 3) & 31;
    int rest = gid >> 8;                             // ot*KSTEPS + t
    int t    = rest % KSTEPS;
    int ot   = rest / KSTEPS;
    int n    = ot * 16 + (lane & 15);
    int f0   = t * 32 + ((lane < 16) ? 0 : 8) + 2 * j + ((j >= 4) ? 8 : 0);
    uint32_t lo = f32_to_bf16_rne(w[(size_t)f0       * C_OUT + n]);
    uint32_t hi = f32_to_bf16_rne(w[(size_t)(f0 + 1) * C_OUT + n]);
    wf[gid] = lo | (hi << 16);
}

// ---------------- Kernel 5: gathered WMMA GEMM + bias + ELU -----------------
// One wave32 computes one 16x16 output tile; K=576 -> 18 x v_wmma_f32_16x16x32_bf16.
// A fragment (16x32 bf16): lane L holds row M=L%16; K offset 0/8 by L>=16;
// the spiral gather is folded into the two 16B A-fragment loads per k-step.
__global__ __launch_bounds__(256) void gemm_kernel(const uint16_t* __restrict__ pooled_bf,
                                                   const uint32_t* __restrict__ wfrag,
                                                   const int*      __restrict__ spiral,
                                                   const float*    __restrict__ bias,
                                                   float*          __restrict__ out) {
    int wave = (blockIdx.x * 256 + (int)threadIdx.x) >> 5;
    int lane = threadIdx.x & 31;
    int ot = wave & 3;                 // C_OUT tile
    int nt = (wave >> 2) % NTILES;     // N_OUT tile
    int b  = (wave >> 2) / NTILES;     // batch
    int m  = lane & 15;
    int hi = lane >> 4;

    // 9 spiral row indices for this lane's output row
    int n_row = nt * 16 + m;
    int rows[S_];
    #pragma unroll
    for (int s = 0; s < S_; ++s) rows[s] = spiral[n_row * S_ + s];

    v8f acc = {};
    #pragma unroll
    for (int t = 0; t < KSTEPS; ++t) {
        int s = t >> 1;                                    // spiral segment
        const uint16_t* rp = pooled_bf
            + ((size_t)b * N_OUT + rows[s]) * C_IN         // gathered row (128B aligned)
            + (t & 1) * 32 + hi * 8;                       // k chunk + lane half
        uint4 a0 = *(const uint4*)rp;                      // K +0..7   (pairs)
        uint4 a1 = *(const uint4*)(rp + 16);               // K +16..23 (pairs)
        const uint4* bp = (const uint4*)(wfrag + (((ot * KSTEPS + t) * 32 + lane) << 3));
        uint4 b0 = bp[0];
        uint4 b1 = bp[1];
        union { uint32_t u[8]; v16bf v; } A, Bf;
        A.u[0] = a0.x; A.u[1] = a0.y; A.u[2] = a0.z; A.u[3] = a0.w;
        A.u[4] = a1.x; A.u[5] = a1.y; A.u[6] = a1.z; A.u[7] = a1.w;
        Bf.u[0] = b0.x; Bf.u[1] = b0.y; Bf.u[2] = b0.z; Bf.u[3] = b0.w;
        Bf.u[4] = b1.x; Bf.u[5] = b1.y; Bf.u[6] = b1.z; Bf.u[7] = b1.w;
        acc = __builtin_amdgcn_wmma_f32_16x16x32_bf16(
            /*neg_a=*/false, A.v, /*neg_b=*/false, Bf.v,
            /*c_mod=*/(short)0, acc, /*reuse_a=*/false, /*reuse_b=*/false);
    }

    // C/D layout: lane L, VGPR v -> (M = v + 8*hi, N = L%16)
    float bv = bias[ot * 16 + m];
    size_t ob = ((size_t)b * N_OUT + (size_t)nt * 16) * C_OUT + ot * 16 + m;
    #pragma unroll
    for (int v = 0; v < 8; ++v) {
        int mr = v + hi * 8;
        float val = acc[v] + bv;
        val = val > 0.f ? val : (__expf(val) - 1.f);       // ELU (alpha=1)
        out[ob + (size_t)mr * C_OUT] = val;
    }
}

extern "C" void kernel_launch(void* const* d_in, const int* in_sizes, int n_in,
                              void* d_out, int out_size, void* d_ws, size_t ws_size,
                              hipStream_t stream) {
    const float* x      = (const float*)d_in[0];
    const float* values = (const float*)d_in[1];
    const float* weight = (const float*)d_in[2];
    const float* bias   = (const float*)d_in[3];
    const int*   row    = (const int*)d_in[4];
    const int*   col    = (const int*)d_in[5];
    const int*   spiral = (const int*)d_in[6];
    float*       out    = (float*)d_out;

    // Workspace layout (all 16B aligned):
    //   [0)                         fp32 pooled  : B*N_OUT*64 floats  (114.7 MB)
    //   [pooled_f32_bytes)          bf16 pooled  : same elems as u16  ( 57.3 MB)
    //   [+pooled_bf16_bytes)        B fragments  : 4*18*32*8 dwords   ( 72  KB)
    const size_t pooled_elems = (size_t)B_ * N_OUT * C_IN;   // 28,672,000
    char* ws = (char*)d_ws;
    float*    pooled_f32 = (float*)ws;
    uint16_t* pooled_bf  = (uint16_t*)(ws + pooled_elems * sizeof(float));
    uint32_t* wfrag      = (uint32_t*)(ws + pooled_elems * (sizeof(float) + sizeof(uint16_t)));

    // 1) zero accumulator (7,168,000 float4 -> 28000 blocks)
    zero_f32_kernel<<<(int)(pooled_elems / 4 / 256), 256, 0, stream>>>(
        (float4*)pooled_f32, (int)(pooled_elems / 4));

    // 2) scatter-add edges (84000*64 threads -> 21000 blocks)
    scatter_kernel<<<(NNZ * 64) / 256, 256, 0, stream>>>(x, values, row, col, pooled_f32);

    // 3) convert pooled to bf16 (14,336,000 pairs -> 56000 blocks)
    cvt_bf16_kernel<<<(int)(pooled_elems / 2 / 256), 256, 0, stream>>>(
        (const float2*)pooled_f32, (uint32_t*)pooled_bf, (int)(pooled_elems / 2));

    // 4) pre-swizzle weight into WMMA B-fragment layout (18432 -> 72 blocks)
    wfrag_kernel<<<(OTILES * KSTEPS * 32 * 8) / 256, 256, 0, stream>>>(weight, wfrag);

    // 5) gathered GEMM: 16*1750*4 = 112000 waves, 8 waves/block -> 14000 blocks
    gemm_kernel<<<(B_ * NTILES * OTILES) / 8, 256, 0, stream>>>(
        pooled_bf, wfrag, spiral, bias, out);
}
